// ChannelAttention_53566832115936
// MI455X (gfx1250) — matrix operand order repaired and checked by reading, not compiled, not run on hardware
//
#include <hip/hip_runtime.h>
#include <hip/hip_bf16.h>
#include <stdint.h>

typedef __attribute__((ext_vector_type(16))) __bf16 v16bf;
typedef __attribute__((ext_vector_type(8)))  float  v8f;

union BF16x16 { uint32_t u[8]; v16bf v; };

__device__ __forceinline__ uint16_t f2bf(float x) {
  uint32_t u = __float_as_uint(x);
  uint32_t r = (u + 0x7FFFu + ((u >> 16) & 1u)) >> 16;
  return (uint16_t)r;
}

__device__ __forceinline__ v8f wmma_bf16(v16bf a, v16bf b, v8f c) {
  return __builtin_amdgcn_wmma_f32_16x16x32_bf16(false, a, false, b, (short)0, c, false, false);
}

// async global->LDS copy of 16 bytes per lane (ASYNCcnt-tracked, no VGPR staging)
__device__ __forceinline__ void async_copy16(const void* gsrc, void* lds_dst) {
  uint32_t loff = (uint32_t)(size_t)lds_dst;  // flat->LDS: low 32 bits are the LDS offset
  asm volatile("global_load_async_to_lds_b128 %0, %1, off"
               :: "v"(loff), "v"(gsrc) : "memory");
}
__device__ __forceinline__ void wait_async0() {
  asm volatile("s_wait_asynccnt 0x0" ::: "memory");
}

// A-matrix frag (16x32 bf16): lane<16 -> row, K in {0..7,16..23}; lane>=16 same rows, K {8..15,24..31}
__device__ __forceinline__ v16bf load_frag_A(const uint32_t* p32, int row, int ldHalf, int kBase, int hi) {
  BF16x16 f;
#pragma unroll
  for (int r = 0; r < 8; ++r) {
    int k = (r < 4 ? 2 * r : 2 * r + 8) + hi * 8 + kBase;
    f.u[r] = p32[row * ldHalf + (k >> 1)];
  }
  return f.v;
}

// B-matrix frag (32x16 bf16) from transposed storage Bt[col][K]: lane<16 -> col, K {0..7,16..23}; lane>=16 -> K {8..15,24..31}
__device__ __forceinline__ v16bf load_frag_B(const uint32_t* p32, int col, int ldHalf, int kBase, int hi) {
  BF16x16 f;
#pragma unroll
  for (int r = 0; r < 8; ++r) {
    int k = (r < 4 ? 2 * r : 2 * r + 8) + hi * 8 + kBase;
    f.u[r] = p32[col * ldHalf + (k >> 1)];
  }
  return f.v;
}

// ---------------- stage 0a: f32 -> bf16 cast (weights) ----------------
__global__ void cast_bf16_kernel(const float* __restrict__ in, uint16_t* __restrict__ out, int n) {
  int i = blockIdx.x * blockDim.x + threadIdx.x;
  if (i < n) out[i] = f2bf(in[i]);
}

// ---------------- stage 0b: x (B,C,N) -> tokens (B,N,C) bf16 ----------------
__global__ __launch_bounds__(256) void x_to_tokens_kernel(const float* __restrict__ x,
                                                          uint16_t* __restrict__ tok) {
  __shared__ float tile[32][33];
  int b = blockIdx.z;
  int n0 = blockIdx.x * 32, c0 = blockIdx.y * 32;
  int tx = threadIdx.x & 31, ty = threadIdx.x >> 5;
#pragma unroll
  for (int i = 0; i < 4; ++i) {
    int c = c0 + ty + i * 8;
    tile[ty + i * 8][tx] = x[((size_t)b * 512 + c) * 4096 + n0 + tx];
  }
  __syncthreads();
#pragma unroll
  for (int i = 0; i < 4; ++i) {
    int n = n0 + ty + i * 8;
    tok[((size_t)b * 4096 + n) * 512 + c0 + tx] = f2bf(tile[tx][ty + i * 8]);
  }
}

// ---------------- generic bf16 WMMA GEMM: acc[m,n] = sum_k A[m,k]*Bt[n,k] ----------------
// Double-buffered async global->LDS staging.
// EPI==0: QKV epilogue (scatter to qT/ks/v, scale k); EPI==1: bias + f32 out
template <int EPI>
__global__ __launch_bounds__(256) void gemm_wmma_kernel(
    const uint16_t* __restrict__ A, const uint16_t* __restrict__ Bt,
    int M, int K,
    uint16_t* __restrict__ qT, uint16_t* __restrict__ ks, uint16_t* __restrict__ vv,
    const float* __restrict__ bias, float* __restrict__ outF) {
  __shared__ uint16_t sA[2][128 * 32];
  __shared__ uint16_t sB[2][64 * 32];
  const int b    = blockIdx.z;
  const int m0   = blockIdx.x * 128;
  const int n0   = blockIdx.y * 64;
  const int tid  = threadIdx.x;
  const int lane = tid & 31, wid = tid >> 5;
  const int hi = lane >> 4, lo16 = lane & 15;
  const int wm = wid >> 1, wn = wid & 1;  // 4 x 2 wave grid, each wave 32x32
  const uint16_t* Ab = A + (size_t)b * M * K;

  // per-thread staging coordinates
  const int arow = tid >> 1, akk = (tid & 1) * 16;       // A tile: 128x32, 16 elems/thread x2
  const int bcol = (tid & 127) >> 1, bkk = (tid & 1) * 16;
  const uint16_t* aSrc = Ab + (size_t)(m0 + arow) * K + akk;
  const uint16_t* bSrc = Bt + (size_t)(n0 + bcol) * K + bkk;

  v8f acc[2][2];
#pragma unroll
  for (int i = 0; i < 2; ++i)
#pragma unroll
    for (int j = 0; j < 2; ++j) acc[i][j] = (v8f){0.f, 0.f, 0.f, 0.f, 0.f, 0.f, 0.f, 0.f};

  const int KB = K >> 5;
  // prologue: issue tile 0 into buffer 0
  async_copy16(aSrc, &sA[0][arow * 32 + akk]);
  async_copy16(aSrc + 8, &sA[0][arow * 32 + akk + 8]);
  if (tid < 128) {
    async_copy16(bSrc, &sB[0][bcol * 32 + bkk]);
    async_copy16(bSrc + 8, &sB[0][bcol * 32 + bkk + 8]);
  }

  for (int kb = 0; kb < KB; ++kb) {
    const int cur = kb & 1;
    wait_async0();
    __syncthreads();
    if (kb + 1 < KB) {  // prefetch next tile into alternate buffer (overlaps compute)
      const int nxt = 1 - cur;
      const int k0 = (kb + 1) * 32;
      async_copy16(aSrc + k0, &sA[nxt][arow * 32 + akk]);
      async_copy16(aSrc + k0 + 8, &sA[nxt][arow * 32 + akk + 8]);
      if (tid < 128) {
        async_copy16(bSrc + k0, &sB[nxt][bcol * 32 + bkk]);
        async_copy16(bSrc + k0 + 8, &sB[nxt][bcol * 32 + bkk + 8]);
      }
    }
    const uint32_t* a32 = (const uint32_t*)sA[cur];
    const uint32_t* b32 = (const uint32_t*)sB[cur];
    v16bf bf[2];
#pragma unroll
    for (int nt = 0; nt < 2; ++nt)
      bf[nt] = load_frag_B(b32, wn * 32 + nt * 16 + lo16, 16, 0, hi);
#pragma unroll
    for (int mt = 0; mt < 2; ++mt) {
      v16bf af = load_frag_A(a32, wm * 32 + mt * 16 + lo16, 16, 0, hi);
#pragma unroll
      for (int nt = 0; nt < 2; ++nt) acc[mt][nt] = wmma_bf16(af, bf[nt], acc[mt][nt]);
    }
    __syncthreads();  // all frag reads done before next iteration's buffer swap
  }

  // epilogue: C-layout -> element (row = g + 8*hi, col = lo16) per 16x16 tile
#pragma unroll
  for (int mt = 0; mt < 2; ++mt)
#pragma unroll
    for (int nt = 0; nt < 2; ++nt)
#pragma unroll
      for (int g = 0; g < 8; ++g) {
        float val = acc[mt][nt][g];
        int rrow = m0 + wm * 32 + mt * 16 + g + 8 * hi;
        int ccol = n0 + wn * 32 + nt * 16 + lo16;
        if (EPI == 0) {
          int which = ccol >> 9, rem = ccol & 511;
          int head = rem >> 6, d = rem & 63;
          size_t bh = (size_t)b * 8 + head;
          if (which == 0)      qT[(bh * 64 + d) * 4096 + rrow] = f2bf(val);
          else if (which == 1) ks[(bh * 4096 + rrow) * 64 + d] = f2bf(val * 0.125f);
          else                 vv[(bh * 4096 + rrow) * 64 + d] = f2bf(val);
        } else {
          outF[((size_t)b * M + rrow) * 512 + ccol] = val + bias[ccol];
        }
      }
}

// ---------------- flash attention: softmax_m(ks . v^T) @ q ----------------
// Double-buffered async staging of v / q^T blocks; out written bf16 in
// scrambled (B, C, N) layout for the faithful-reshape projection.
__global__ __launch_bounds__(256) void flash_attn_kernel(
    const uint16_t* __restrict__ ks, const uint16_t* __restrict__ vv,
    const uint16_t* __restrict__ qT, uint16_t* __restrict__ Obuf) {
  __shared__ uint16_t sV[2][64 * 64];
  __shared__ uint16_t sQ[2][64 * 64];
  __shared__ uint16_t sP[8 * 16 * 64];
  const int bh = blockIdx.y;                 // 0..15
  const int tid = threadIdx.x;
  const int lane = tid & 31, wid = tid >> 5;
  const int hi = lane >> 4, lo16 = lane & 15;
  const int r0 = blockIdx.x * 128 + wid * 16;  // wave's 16 rows

  // per-thread staging coordinates (16 bf16 = 32B per thread per array, as 2 async b128s)
  const int srow = tid >> 2, soff = (tid & 3) * 16;
  const uint16_t* vSrc = vv + (size_t)bh * 4096 * 64 + (size_t)srow * 64 + soff;
  const uint16_t* qSrc = qT + ((size_t)bh * 64 + srow) * 4096 + soff;

  // resident A-frags of ks rows (K = d = 64)
  const uint32_t* ksg = (const uint32_t*)(ks + (size_t)bh * 4096 * 64);
  v16bf aK0 = load_frag_A(ksg, r0 + lo16, 32, 0, hi);
  v16bf aK1 = load_frag_A(ksg, r0 + lo16, 32, 32, hi);

  v8f o[4];
#pragma unroll
  for (int dt = 0; dt < 4; ++dt) o[dt] = (v8f){0.f, 0.f, 0.f, 0.f, 0.f, 0.f, 0.f, 0.f};
  float mrun[8], lrun[8];
#pragma unroll
  for (int g = 0; g < 8; ++g) { mrun[g] = -1e30f; lrun[g] = 0.f; }

  // prologue: issue block j=0 into buffer 0 (v block j covers rows j*64.., q^T cols j*64..)
  async_copy16(vSrc, &sV[0][srow * 64 + soff]);
  async_copy16(vSrc + 8, &sV[0][srow * 64 + soff + 8]);
  async_copy16(qSrc, &sQ[0][srow * 64 + soff]);
  async_copy16(qSrc + 8, &sQ[0][srow * 64 + soff + 8]);

  for (int j = 0; j < 64; ++j) {
    const int cur = j & 1;
    wait_async0();
    __syncthreads();
    if (j + 1 < 64) {  // prefetch next block into alternate buffer (overlaps compute)
      const int nxt = 1 - cur;
      const size_t vstep = (size_t)(j + 1) * 64 * 64;  // v: advance 64 rows
      const size_t qstep = (size_t)(j + 1) * 64;       // qT: advance 64 cols
      async_copy16(vSrc + vstep, &sV[nxt][srow * 64 + soff]);
      async_copy16(vSrc + vstep + 8, &sV[nxt][srow * 64 + soff + 8]);
      async_copy16(qSrc + qstep, &sQ[nxt][srow * 64 + soff]);
      async_copy16(qSrc + qstep + 8, &sQ[nxt][srow * 64 + soff + 8]);
    }
    const uint32_t* v32 = (const uint32_t*)sV[cur];
    const uint32_t* q32 = (const uint32_t*)sQ[cur];

    v8f st[4];
#pragma unroll
    for (int nt = 0; nt < 4; ++nt) {
      v16bf b0 = load_frag_B(v32, nt * 16 + lo16, 32, 0, hi);
      v16bf b1 = load_frag_B(v32, nt * 16 + lo16, 32, 32, hi);
      v8f z = (v8f){0.f, 0.f, 0.f, 0.f, 0.f, 0.f, 0.f, 0.f};
      st[nt] = wmma_bf16(aK1, b1, wmma_bf16(aK0, b0, z));
    }

    float mnew[8], scale[8];
#pragma unroll
    for (int g = 0; g < 8; ++g) {
      float mx = fmaxf(fmaxf(st[0][g], st[1][g]), fmaxf(st[2][g], st[3][g]));
#pragma unroll
      for (int s = 1; s < 16; s <<= 1) mx = fmaxf(mx, __shfl_xor(mx, s, 32));
      mnew[g] = fmaxf(mrun[g], mx);
      scale[g] = __expf(mrun[g] - mnew[g]);
      mrun[g] = mnew[g];
    }
#pragma unroll
    for (int nt = 0; nt < 4; ++nt)
#pragma unroll
      for (int g = 0; g < 8; ++g) st[nt][g] = __expf(st[nt][g] - mnew[g]);
#pragma unroll
    for (int g = 0; g < 8; ++g) {
      float rs = st[0][g] + st[1][g] + st[2][g] + st[3][g];
#pragma unroll
      for (int s = 1; s < 16; s <<= 1) rs += __shfl_xor(rs, s, 32);
      lrun[g] = lrun[g] * scale[g] + rs;
    }
    // P -> per-wave LDS tile (C-layout to A-layout relayout)
    uint16_t* pbase = sP + wid * 16 * 64;
#pragma unroll
    for (int nt = 0; nt < 4; ++nt)
#pragma unroll
      for (int g = 0; g < 8; ++g)
        pbase[(g + 8 * hi) * 64 + nt * 16 + lo16] = f2bf(st[nt][g]);
    // rescale running accumulator
#pragma unroll
    for (int dt = 0; dt < 4; ++dt)
#pragma unroll
      for (int g = 0; g < 8; ++g) o[dt][g] *= scale[g];
    __syncthreads();

    const uint32_t* p32 = (const uint32_t*)pbase;
    v16bf aP0 = load_frag_A(p32, lo16, 32, 0, hi);
    v16bf aP1 = load_frag_A(p32, lo16, 32, 32, hi);
#pragma unroll
    for (int dt = 0; dt < 4; ++dt) {
      v16bf b0 = load_frag_B(q32, dt * 16 + lo16, 32, 0, hi);
      v16bf b1 = load_frag_B(q32, dt * 16 + lo16, 32, 32, hi);
      o[dt] = wmma_bf16(aP1, b1, wmma_bf16(aP0, b0, o[dt]));
    }
  }

  int bb = bh >> 3, hh = bh & 7;
#pragma unroll
  for (int g = 0; g < 8; ++g) {
    float inv = 1.0f / lrun[g];
    int n = r0 + g + 8 * hi;
#pragma unroll
    for (int dt = 0; dt < 4; ++dt) {
      int d = dt * 16 + lo16;
      Obuf[((size_t)bb * 512 + hh * 64 + d) * 4096 + n] = f2bf(o[dt][g] * inv);
    }
  }
}

extern "C" void kernel_launch(void* const* d_in, const int* in_sizes, int n_in,
                              void* d_out, int out_size, void* d_ws, size_t ws_size,
                              hipStream_t stream) {
  (void)in_sizes; (void)n_in; (void)out_size; (void)ws_size;
  const float* x      = (const float*)d_in[0];  // (2,512,64,64)
  const float* w_qkv  = (const float*)d_in[1];  // (1536,512)
  const float* w_proj = (const float*)d_in[2];  // (512,512)
  const float* b_proj = (const float*)d_in[3];  // (512,)
  float* out = (float*)d_out;                   // (2,512,64,64)

  char* ws = (char*)d_ws;
  size_t off = 0;
  auto take = [&](size_t bytes) -> char* {
    char* p = ws + off;
    off += (bytes + 255) & ~size_t(255);
    return p;
  };
  uint16_t* tok   = (uint16_t*)take(2ull * 4096 * 512 * 2);   // tokens bf16 (B,N,C)
  uint16_t* wqkb  = (uint16_t*)take(1536ull * 512 * 2);       // w_qkv bf16
  uint16_t* wprb  = (uint16_t*)take(512ull * 512 * 2);        // w_proj bf16
  uint16_t* ksb   = (uint16_t*)take(16ull * 4096 * 64 * 2);   // k*scale (bh,n,d)
  uint16_t* vb    = (uint16_t*)take(16ull * 4096 * 64 * 2);   // v (bh,n,d)
  uint16_t* qTb   = (uint16_t*)take(16ull * 64 * 4096 * 2);   // q^T (bh,d,n)
  uint16_t* Ob    = (uint16_t*)take(2ull * 512 * 4096 * 2);   // attn out, scrambled (B,C,N)

  // stage 0: casts + token transpose
  cast_bf16_kernel<<<(1536 * 512 + 255) / 256, 256, 0, stream>>>(w_qkv, wqkb, 1536 * 512);
  cast_bf16_kernel<<<(512 * 512 + 255) / 256, 256, 0, stream>>>(w_proj, wprb, 512 * 512);
  x_to_tokens_kernel<<<dim3(128, 16, 2), 256, 0, stream>>>(x, tok);

  // stage 1: QKV GEMM (M=4096, N=1536, K=512) per batch
  gemm_wmma_kernel<0><<<dim3(32, 24, 2), 256, 0, stream>>>(
      tok, wqkb, 4096, 512, qTb, ksb, vb, nullptr, nullptr);

  // stage 2: flash attention over 16 (b,h) problems, 32 row-blocks each
  flash_attn_kernel<<<dim3(32, 16), 256, 0, stream>>>(ksb, vb, qTb, Ob);

  // stage 3: projection GEMM on scrambled linear view (M=4096, N=512, K=512) per batch + bias
  gemm_wmma_kernel<1><<<dim3(32, 8, 2), 256, 0, stream>>>(
      Ob, wprb, 4096, 512, nullptr, nullptr, nullptr, b_proj, out);
}
